// Model_19602230739101
// MI455X (gfx1250) — matrix operand order repaired and checked by reading, not compiled, not run on hardware
//
#include <hip/hip_runtime.h>

// ---------------------------------------------------------------------------
// CDNA5 (gfx1250) implementation: all dense math via v_wmma_f32_16x16x32_f16,
// f32 accumulation, f16 activation storage. Wave32 everywhere.
// Each GEMM wave accumulates NSUB (=4) 16x16 output tiles per A-fragment to
// amortize A loads (10 b128 loads per 4 WMMAs instead of 16).
// ---------------------------------------------------------------------------

typedef __attribute__((ext_vector_type(16))) _Float16 v16h;
typedef __attribute__((ext_vector_type(8)))  _Float16 v8h;
typedef __attribute__((ext_vector_type(8)))  float    v8f;

static constexpr int  HH  = 32;
static constexpr int  BB  = 2048;
static constexpr int  LLx = 256;
static constexpr long HB  = (long)HH * BB;   // 65536 sample rows
static constexpr long R11 = HB * 11;         // 720896 stacked state rows

#define DEV __device__ __forceinline__

DEV float sigf(float x) { return 1.0f / (1.0f + __expf(-x)); }

// ---------------------------------------------------------------------------
// WMMA GEMM: C[M x (NT*16)] = A(f16, row-major, lda mult of 32) x Bp (+ bias)
// One wave computes NSUB adjacent 16x16 tiles (grid.y = NT/NSUB).
// A-fragment per ISA 16-bit A 16x32 layout:
//   lane = 16*h + m ; halves [8h,8h+8) and [16+8h,16+8h+8) of the k-slice.
// B packed as [kt][nt][lane][16] so each lane reads 32 contiguous bytes.
// C/D layout: acc[r] = C[mtile*16 + r + 8h][nt*16 + (lane&15)].
// ---------------------------------------------------------------------------
template <typename OutT, int NSUB>
__global__ __launch_bounds__(32)
void gemm_wmma_k(const _Float16* __restrict__ A, int lda,
                 const _Float16* __restrict__ Bp, const float* __restrict__ bias,
                 int Nreal, OutT* __restrict__ C, int ldc, int KT, int NT)
{
    const int lane = threadIdx.x;
    const int h  = lane >> 4;
    const int ml = lane & 15;
    const size_t mt = blockIdx.x;
    const int ntBase = blockIdx.y * NSUB;

    v8f acc[NSUB];
#pragma unroll
    for (int i = 0; i < NSUB; ++i) {
        int n = (ntBase + i) * 16 + ml;
        float bv = (bias != nullptr && n < Nreal) ? bias[n] : 0.0f;
        acc[i] = (v8f){ bv, bv, bv, bv, bv, bv, bv, bv };
    }

    const _Float16* ap = A + (mt * 16 + (size_t)ml) * (size_t)lda + 8 * h;
    const _Float16* bp = Bp + ((size_t)ntBase * 32 + lane) * 16;
    const size_t bstep = (size_t)NT * 512;

    for (int kt = 0; kt < KT; ++kt) {
        v8h lo = *(const v8h*)(ap);
        v8h hi = *(const v8h*)(ap + 16);
        if (kt + 1 < KT) __builtin_prefetch(bp + bstep, 0, 1);  // global_prefetch
        v16h af;
#pragma unroll
        for (int i = 0; i < 8; ++i) { af[i] = lo[i]; af[8 + i] = hi[i]; }
#pragma unroll
        for (int i = 0; i < NSUB; ++i) {
            v16h bf = *(const v16h*)(bp + (size_t)i * 512);
            acc[i] = __builtin_amdgcn_wmma_f32_16x16x32_f16(false, af, false, bf,
                                                            (short)0, acc[i], false, false);
        }
        ap += 32;
        bp += bstep;
    }

#pragma unroll
    for (int i = 0; i < NSUB; ++i) {
        OutT* cp = C + (mt * 16 + (size_t)(8 * h)) * (size_t)ldc + (ntBase + i) * 16 + ml;
#pragma unroll
        for (int r = 0; r < 8; ++r) cp[(size_t)r * ldc] = (OutT)acc[i][r];
    }
}

// ---------------------------------------------------------------------------
// Weight packer: builds the per-lane B fragment layout from up to 3 stacked
// f32 (K x N) row-major sources occupying consecutive slotK-row K-slots.
// Pads K and N with zeros.
// ---------------------------------------------------------------------------
__global__ void pack_w_k(const float* __restrict__ s0, int k0,
                         const float* __restrict__ s1, int k1,
                         const float* __restrict__ s2, int k2,
                         int slotK, int Nreal, int NT, int KT,
                         _Float16* __restrict__ out)
{
    long idx = (long)blockIdx.x * blockDim.x + threadIdx.x;
    long total = (long)KT * NT * 512;
    if (idx >= total) return;
    int j    = (int)(idx & 15);
    int lane = (int)((idx >> 4) & 31);
    long tile = idx >> 9;
    int nt = (int)(tile % NT);
    int kt = (int)(tile / NT);
    int h = lane >> 4, ml = lane & 15;
    int n = nt * 16 + ml;
    int k = kt * 32 + (j < 8 ? 8 * h + j : 16 + 8 * h + (j - 8));
    float v = 0.0f;
    if (n < Nreal) {
        int slot = k / slotK, kl = k % slotK;
        const float* src = (slot == 0) ? s0 : (slot == 1) ? s1 : s2;
        int kr           = (slot == 0) ? k0 : (slot == 1) ? k1 : k2;
        if (src != nullptr && kl < kr) v = src[(size_t)kl * Nreal + n];
    }
    out[idx] = (_Float16)v;
}

// ---------------------------------------------------------------------------
// LayerNorm (+ optional per-row-slot pre-bias) + ReLU, f32 in -> f16 out.
// One wave per row. bias_mode: 0 = bA for all rows, 1 = slot r%3 -> bA/bB/bC,
// 2 = (r%11==0) ? bA : bB. Output row r -> out + (r/ogroup)*ostride
//                                       + (r%ogroup)*oslot ; cols [C,padC)=0.
// ---------------------------------------------------------------------------
__global__ void ln_relu_k(const float* __restrict__ in, int istride, int C, int padC,
                          const float* __restrict__ g, const float* __restrict__ bt,
                          const float* __restrict__ bA, const float* __restrict__ bB,
                          const float* __restrict__ bC, int bias_mode,
                          _Float16* __restrict__ out, int ogroup, int oslot,
                          long ostride, long rows)
{
    long wid = (long)blockIdx.x * (blockDim.x >> 5) + (threadIdx.x >> 5);
    int lane = threadIdx.x & 31;
    if (wid >= rows) return;
    const float* bias = bA;
    if (bias_mode == 1) { int s = (int)(wid % 3); bias = (s == 0) ? bA : (s == 1) ? bB : bC; }
    else if (bias_mode == 2) { bias = ((wid % 11) == 0) ? bA : bB; }

    const float* x = in + (size_t)wid * istride;
    float s = 0.0f;
    for (int c = lane; c < C; c += 32) s += x[c] + (bias ? bias[c] : 0.0f);
    for (int o = 16; o > 0; o >>= 1) s += __shfl_xor(s, o, 32);
    float mean = s / (float)C;
    float v = 0.0f;
    for (int c = lane; c < C; c += 32) {
        float d = x[c] + (bias ? bias[c] : 0.0f) - mean;
        v += d * d;
    }
    for (int o = 16; o > 0; o >>= 1) v += __shfl_xor(v, o, 32);
    float rsq = rsqrtf(v / (float)C + 1e-5f);

    _Float16* op = out + (wid / ogroup) * ostride + (long)(wid % ogroup) * oslot;
    for (int c = lane; c < padC; c += 32) {
        float y = 0.0f;
        if (c < C) {
            y = (x[c] + (bias ? bias[c] : 0.0f) - mean) * rsq * g[c] + bt[c];
            y = y > 0.0f ? y : 0.0f;
        }
        op[c] = (_Float16)y;
    }
}

// --------------------------- small utility kernels -------------------------
__global__ void zero_f16_k(_Float16* p, long n)
{
    long i = (long)blockIdx.x * blockDim.x + threadIdx.x;
    if (i < n) p[i] = (_Float16)0.0f;
}

__global__ void cvt_f32_f16_k(const float* __restrict__ s, _Float16* __restrict__ d, long n)
{
    long i = (long)blockIdx.x * blockDim.x + threadIdx.x;
    if (i < n) d[i] = (_Float16)s[i];
}

__global__ void copy_f32_k(float* __restrict__ d, const float* __restrict__ s, long n)
{
    long i = (long)blockIdx.x * blockDim.x + threadIdx.x;
    if (i < n) d[i] = s[i];
}

__global__ void bias2_k(const float* a, const float* b, float* o, int n)
{
    int i = blockIdx.x * blockDim.x + threadIdx.x;
    if (i < n) o[i] = a[i] + b[i];
}

// SIN (HB*3 x 96) rows: [ms | psit | ball] in K-slots 0/32/64.
__global__ void fill_sin_k(const float* __restrict__ ms, const float* __restrict__ ps,
                           const float* __restrict__ bs, _Float16* __restrict__ out)
{
    long idx = (long)blockIdx.x * blockDim.x + threadIdx.x;
    if (idx >= HB * 48) return;
    long s = idx / 48;
    int  j = (int)(idx % 48);
    if (j < 11)      out[(s * 3 + 0) * 96 +      j      ] = (_Float16)ms[s * 11 + j];
    else if (j < 30) out[(s * 3 + 1) * 96 + 32 + (j - 11)] = (_Float16)ps[s * 19 + (j - 11)];
    else             out[(s * 3 + 2) * 96 + 64 + (j - 30)] = (_Float16)bs[s * 18 + (j - 30)];
}

// PIN/OIN (R11 x 64) rows: row s*11   = "one" state (cols 0..28),
//                          row s*11+i = team state i-1 (cols 32..60).
__global__ void fill_pair_k(const float* __restrict__ one, const float* __restrict__ team,
                            _Float16* __restrict__ out)
{
    long idx = (long)blockIdx.x * blockDim.x + threadIdx.x;
    if (idx >= R11 * 29) return;
    long r = idx / 29;
    int  k = (int)(idx % 29);
    long s = r / 11;
    int it = (int)(r % 11);
    float v; int col;
    if (it == 0) { v = one[s * 29 + k]; col = k; }
    else         { v = team[(s * 10 + (it - 1)) * 29 + k]; col = 32 + k; }
    out[r * 64 + col] = (_Float16)v;
}

// ---------------------------------------------------------------------------
// Attention stage part 1 (one wave / sample): rq/lk/rv/lv have row stride 64.
//   att1 = softmax_k(rq1 @ lk1^T)  (11x11), emb = att1^T @ rv1,
//   sub  = lv1 - emb  -> f16 (stride 64, cols 48..63 zeroed)
// ---------------------------------------------------------------------------
__global__ void att_stage1_k(const _Float16* __restrict__ rq, const _Float16* __restrict__ lk,
                             const _Float16* __restrict__ rv, const _Float16* __restrict__ lv,
                             _Float16* __restrict__ sub, float* __restrict__ att1_out)
{
    const long s = blockIdx.x;
    const int lane = threadIdx.x;
    __shared__ float att[121];

    const _Float16* qm = rq + s * 11 * 64;
    const _Float16* km = lk + s * 11 * 64;
    const _Float16* vm = rv + s * 11 * 64;
    const _Float16* lm = lv + s * 11 * 64;

    for (int p = lane; p < 121; p += 32) {
        int qi = p / 11, ki = p % 11;
        float acc = 0.0f;
        for (int d = 0; d < 48; ++d)
            acc += (float)qm[qi * 64 + d] * (float)km[ki * 64 + d];
        att[p] = acc;
    }
    __syncthreads();

    if (lane < 11) {
        int qi = lane;
        float mx = -1e30f;
        for (int k = 0; k < 11; ++k) mx = fmaxf(mx, att[qi * 11 + k]);
        float sm = 0.0f;
        for (int k = 0; k < 11; ++k) { float e = __expf(att[qi * 11 + k] - mx); att[qi * 11 + k] = e; sm += e; }
        float inv = 1.0f / sm;
        for (int k = 0; k < 11; ++k) {
            att[qi * 11 + k] *= inv;
            att1_out[s * 121 + qi * 11 + k] = att[qi * 11 + k];
        }
    }
    __syncthreads();

    for (int idx = lane; idx < 11 * 64; idx += 32) {
        int ki = idx / 64, d = idx % 64;
        float val = 0.0f;
        if (d < 48) {
            float e = 0.0f;
            for (int qi = 0; qi < 11; ++qi)
                e += att[qi * 11 + ki] * (float)vm[qi * 64 + d];
            val = (float)lm[ki * 64 + d] - e;
        }
        sub[(s * 11 + ki) * 64 + d] = (_Float16)val;
    }
}

// ---------------------------------------------------------------------------
// Attention stage part 2 (one wave / sample): sq/sk/sv row stride 64.
//   att2 = softmax(sq2[0] . sk2[1..10]) ; agg = att2 @ sv2[1..10] ; pr = sv2[0]
//   writes pr/agg into the f16 concat buffer slices + att2 to d_out.
// ---------------------------------------------------------------------------
__global__ void att_stage2_k(const _Float16* __restrict__ sq, const _Float16* __restrict__ sk,
                             const _Float16* __restrict__ sv, _Float16* __restrict__ catA,
                             int prOff, int agOff, float* __restrict__ att2_out)
{
    const long s = blockIdx.x;
    const int lane = threadIdx.x;
    __shared__ float a2[10];

    const _Float16* q0 = sq + s * 11 * 64;          // row 0
    const _Float16* ks = sk + s * 11 * 64;
    const _Float16* vs = sv + s * 11 * 64;

    if (lane < 10) {
        float acc = 0.0f;
        for (int d = 0; d < 48; ++d)
            acc += (float)q0[d] * (float)ks[(1 + lane) * 64 + d];
        a2[lane] = acc;
    }
    __syncthreads();
    if (lane == 0) {
        float mx = -1e30f;
        for (int k = 0; k < 10; ++k) mx = fmaxf(mx, a2[k]);
        float sm = 0.0f;
        for (int k = 0; k < 10; ++k) { float e = __expf(a2[k] - mx); a2[k] = e; sm += e; }
        float inv = 1.0f / sm;
        for (int k = 0; k < 10; ++k) a2[k] *= inv;
    }
    __syncthreads();
    if (lane < 10) att2_out[s * 10 + lane] = a2[lane];

    _Float16* cp = catA + s * 352;
    for (int d = lane; d < 48; d += 32) {
        float agg = 0.0f;
        for (int k = 0; k < 10; ++k) agg += a2[k] * (float)vs[(1 + k) * 64 + d];
        cp[agOff + d] = (_Float16)agg;
        cp[prOff + d] = vs[d];                       // pr = p_sub @ v2 (row 0)
    }
}

// ------------------------------ LSTM kernels -------------------------------
__global__ void lstm_init_k(const float* __restrict__ h0, const float* __restrict__ c0,
                            _Float16* __restrict__ xh, float* __restrict__ c)
{
    int idx = blockIdx.x * blockDim.x + threadIdx.x;
    if (idx >= BB * LLx) return;
    int b = idx >> 8, j = idx & 255;
    xh[(size_t)b * 512 + 256 + j] = (_Float16)h0[idx];
    c[idx] = c0[idx];
}

__global__ void lstm_prep_k(const _Float16* __restrict__ x, _Float16* __restrict__ xh, int t)
{
    int idx = blockIdx.x * blockDim.x + threadIdx.x;
    if (idx >= BB * LLx) return;
    int b = idx >> 8, j = idx & 255;
    xh[(size_t)b * 512 + j] = x[((size_t)t * BB + b) * 256 + j];
}

__global__ void lstm_cell_k(const float* __restrict__ gt, float* __restrict__ c,
                            float* __restrict__ outb, _Float16* __restrict__ xh, int t)
{
    int idx = blockIdx.x * blockDim.x + threadIdx.x;
    if (idx >= BB * LLx) return;
    int b = idx >> 8, j = idx & 255;
    const float* gr = gt + (size_t)b * 1024;
    float ig = gr[j], fg = gr[256 + j], gg = gr[512 + j], og = gr[768 + j];
    float cn = sigf(fg) * c[idx] + sigf(ig) * tanhf(gg);
    float hn = sigf(og) * tanhf(cn);
    c[idx] = cn;
    outb[((size_t)t * BB + b) * 256 + j] = hn;
    xh[(size_t)b * 512 + 256 + j] = (_Float16)hn;
}

// ------------------------------- head kernels ------------------------------
__global__ void softmax_k(const float* __restrict__ lg, int ld, int n,
                          const float* __restrict__ av, int lda,
                          float* __restrict__ o, int ldo, long rows)
{
    long r = (long)blockIdx.x * blockDim.x + threadIdx.x;
    if (r >= rows) return;
    float buf[12];
    float mx = -1e30f;
    for (int j = 0; j < n; ++j) {
        float x = lg[r * ld + j];
        if (av) x += (av[r * lda + j] - 1.0f) * 1e7f;
        buf[j] = x;
        mx = fmaxf(mx, x);
    }
    float sm = 0.0f;
    for (int j = 0; j < n; ++j) { float e = __expf(buf[j] - mx); buf[j] = e; sm += e; }
    float inv = 1.0f / sm;
    for (int j = 0; j < n; ++j) o[r * ldo + j] = buf[j] * inv;
}

__global__ void dotv_k(const _Float16* __restrict__ a, const float* __restrict__ w,
                       float* __restrict__ o, long rows)
{
    long r = (long)blockIdx.x * blockDim.x + threadIdx.x;
    if (r >= rows) return;
    float acc = 0.0f;
    for (int k = 0; k < 164; ++k) acc += (float)a[r * 192 + k] * w[k];
    o[r] = acc;
}

// ---------------------------------------------------------------------------
extern "C" void kernel_launch(void* const* d_in, const int* in_sizes, int n_in,
                              void* d_out, int out_size, void* d_ws, size_t ws_size,
                              hipStream_t stream)
{
    (void)in_sizes; (void)n_in; (void)out_size; (void)ws_size;

    // ---- inputs (setup_inputs dict order) ----
    const float* in_ms  = (const float*)d_in[0];
    const float* in_ps  = (const float*)d_in[1];
    const float* in_bs  = (const float*)d_in[2];
    const float* in_pst = (const float*)d_in[3];
    const float* in_ost = (const float*)d_in[4];
    const float* in_lst = (const float*)d_in[5];
    const float* in_rst = (const float*)d_in[6];
    const float* in_av  = (const float*)d_in[7];
    const float* in_h0  = (const float*)d_in[8];
    const float* in_c0  = (const float*)d_in[9];
#define PAR(i) ((const float*)d_in[10 + (i)])
    const float *w_match=PAR(0), *b_match=PAR(1), *w_psit=PAR(2), *b_psit=PAR(3),
                *w_ball=PAR(4), *b_ball=PAR(5), *w_pst=PAR(6), *b_pst=PAR(7),
                *w_ost=PAR(8), *b_ost=PAR(9), *w_lst=PAR(10), *b_lst=PAR(11),
                *w_rst=PAR(12), *b_rst=PAR(13);
    const float *q1a=PAR(14),*k1a=PAR(15),*v1a=PAR(16),*q2a=PAR(17),*k2a=PAR(18),*v2a=PAR(19),
                *q1d=PAR(20),*k1d=PAR(21),*v1d=PAR(22),*q2d=PAR(23),*k2d=PAR(24),*v2d=PAR(25);
    const float *w_cat=PAR(26), *b_cat=PAR(27), *g_sit=PAR(28), *bt_sit=PAR(29),
                *g_st=PAR(30), *bt_st=PAR(31), *g_cat=PAR(32), *bt_cat=PAR(33),
                *w_ih=PAR(34), *w_hh=PAR(35), *b_ih=PAR(36), *b_hh=PAR(37),
                *w_a1=PAR(38), *b_a1=PAR(39), *g_a1=PAR(40), *bt_a1=PAR(41),
                *w_a2=PAR(42), *b_a2=PAR(43), *w_m1=PAR(44), *b_m1=PAR(45),
                *g_m1=PAR(46), *bt_m1=PAR(47), *w_m2=PAR(48), *b_m2=PAR(49),
                *w_v1=PAR(50), *b_v1=PAR(51), *g_v1=PAR(52), *bt_v1=PAR(53),
                *w_v2=PAR(54);
#undef PAR

    float* dout = (float*)d_out;
    // output offsets (floats), reference tuple order
    const long O_PROB = 0;
    const long O_PROBM = O_PROB + HB * 12;
    const long O_V    = O_PROBM + HB * 8;
    const long O_HT   = O_V + HB;
    const long O_CT   = O_HT + (long)BB * 256;
    const long O_A2   = O_CT + (long)BB * 256;
    const long O_A1   = O_A2 + HB * 10;
    const long O_D2   = O_A1 + HB * 121;
    const long O_D1   = O_D2 + HB * 10;

    // ---- workspace bump allocator ----
    char* base = (char*)d_ws;
    size_t off = 0;
    auto alloc = [&](size_t bytes) -> char* {
        off = (off + 255) & ~(size_t)255;
        char* p = base + off;
        off += bytes;
        return p;
    };

    _Float16* pk_sit  = (_Float16*)alloc((size_t)3 * 4 * 512 * 2);
    _Float16* pk_stP  = (_Float16*)alloc((size_t)2 * 4 * 512 * 2);
    _Float16* pk_stO  = (_Float16*)alloc((size_t)2 * 4 * 512 * 2);
    _Float16* pk_q[12];
    for (int i = 0; i < 12; ++i) pk_q[i] = (_Float16*)alloc((size_t)2 * 4 * 512 * 2);
    _Float16* pk_cat  = (_Float16*)alloc((size_t)11 * 16 * 512 * 2);
    _Float16* pk_lstm = (_Float16*)alloc((size_t)16 * 64 * 512 * 2);
    _Float16* pk_a1   = (_Float16*)alloc((size_t)8 * 12 * 512 * 2);
    _Float16* pk_m1   = (_Float16*)alloc((size_t)8 * 12 * 512 * 2);
    _Float16* pk_v1   = (_Float16*)alloc((size_t)8 * 12 * 512 * 2);
    _Float16* pk_a2   = (_Float16*)alloc((size_t)6 * 1 * 512 * 2);
    _Float16* pk_m2   = (_Float16*)alloc((size_t)6 * 1 * 512 * 2);
    float*    bsum    = (float*)alloc(1024 * 4);

    _Float16* SINb = (_Float16*)alloc((size_t)HB * 3 * 96 * 2);
    _Float16* PIN  = (_Float16*)alloc((size_t)R11 * 64 * 2);  // becomes P (post-LN)
    _Float16* OIN  = (_Float16*)alloc((size_t)R11 * 64 * 2);  // becomes O (post-LN)
    _Float16* PROJ = (_Float16*)alloc((size_t)4 * R11 * 64 * 2);
    _Float16* SUBB = (_Float16*)alloc((size_t)R11 * 64 * 2);
    _Float16* CATA = (_Float16*)alloc((size_t)HB * 352 * 2);
    _Float16* XB   = (_Float16*)alloc((size_t)HB * 256 * 2);
    _Float16* XH   = (_Float16*)alloc((size_t)BB * 512 * 2);
    float*    CBUF = (float*)alloc((size_t)BB * 256 * 4);
    float*    OUTB = (float*)alloc((size_t)HB * 256 * 4);
    _Float16* OUTH = (_Float16*)alloc((size_t)HB * 256 * 2);
    _Float16* AF   = (_Float16*)alloc((size_t)HB * 192 * 2);

    // f32 GEMM scratch aliases PROJ (disjoint lifetimes):
    //   max f32 temp = R11*64*4 B == exactly 2 of the 4 proj sub-buffers.
    float* F32T = (float*)PROJ;
    _Float16* PR0 = PROJ;
    _Float16* PR1 = PROJ + (size_t)R11 * 64;
    _Float16* PR2 = PROJ + (size_t)2 * R11 * 64;
    _Float16* PR3 = PROJ + (size_t)3 * R11 * 64;

    auto cdiv = [](long a, long b) { return (unsigned)((a + b - 1) / b); };

    auto pack = [&](const float* s0, int k0, const float* s1, int k1,
                    const float* s2, int k2, int slotK, int Nreal, int NT, int KT,
                    _Float16* out) {
        long tot = (long)KT * NT * 512;
        pack_w_k<<<cdiv(tot, 256), 256, 0, stream>>>(s0, k0, s1, k1, s2, k2,
                                                     slotK, Nreal, NT, KT, out);
    };
    auto gemmF4 = [&](const _Float16* A, int lda, const _Float16* Bp, const float* bias,
                      int Nreal, float* C, int ldc, long M, int KT, int NT) {
        dim3 g((unsigned)(M / 16), NT / 4);
        gemm_wmma_k<float, 4><<<g, 32, 0, stream>>>(A, lda, Bp, bias, Nreal, C, ldc, KT, NT);
    };
    auto gemmF1 = [&](const _Float16* A, int lda, const _Float16* Bp, const float* bias,
                      int Nreal, float* C, int ldc, long M, int KT, int NT) {
        dim3 g((unsigned)(M / 16), NT);
        gemm_wmma_k<float, 1><<<g, 32, 0, stream>>>(A, lda, Bp, bias, Nreal, C, ldc, KT, NT);
    };
    auto gemmH4 = [&](const _Float16* A, int lda, const _Float16* Bp, const float* bias,
                      int Nreal, _Float16* C, int ldc, long M, int KT, int NT) {
        dim3 g((unsigned)(M / 16), NT / 4);
        gemm_wmma_k<_Float16, 4><<<g, 32, 0, stream>>>(A, lda, Bp, bias, Nreal, C, ldc, KT, NT);
    };
    auto ln = [&](const float* in, int istride, int C, int padC,
                  const float* g, const float* bt, const float* bA, const float* bB,
                  const float* bC, int mode, _Float16* out, int ogroup, int oslot,
                  long ostride, long rows) {
        ln_relu_k<<<cdiv(rows, 8), 256, 0, stream>>>(in, istride, C, padC, g, bt,
                                                     bA, bB, bC, mode, out, ogroup,
                                                     oslot, ostride, rows);
    };

    // ---- 1. pack all weights into WMMA B-fragment layout (f16) ----
    pack(w_match, 11, w_psit, 19, w_ball, 18, 32, 48, 4, 3, pk_sit);
    pack(w_pst, 29, w_lst, 29, nullptr, 0, 32, 48, 4, 2, pk_stP);
    pack(w_ost, 29, w_rst, 29, nullptr, 0, 32, 48, 4, 2, pk_stO);
    const float* qw[12] = { q1a, k1a, v1a, q2a, k2a, v2a, q1d, k1d, v1d, q2d, k2d, v2d };
    for (int i = 0; i < 12; ++i) pack(qw[i], 48, nullptr, 0, nullptr, 0, 64, 48, 4, 2, pk_q[i]);
    pack(w_cat, 336, nullptr, 0, nullptr, 0, 352, 256, 16, 11, pk_cat);
    pack(w_ih, 256, w_hh, 256, nullptr, 0, 256, 1024, 64, 16, pk_lstm);
    pack(w_a1, 256, nullptr, 0, nullptr, 0, 256, 164, 12, 8, pk_a1);
    pack(w_m1, 256, nullptr, 0, nullptr, 0, 256, 164, 12, 8, pk_m1);
    pack(w_v1, 256, nullptr, 0, nullptr, 0, 256, 164, 12, 8, pk_v1);
    pack(w_a2, 164, nullptr, 0, nullptr, 0, 192, 12, 1, 6, pk_a2);
    pack(w_m2, 164, nullptr, 0, nullptr, 0, 192, 8, 1, 6, pk_m2);
    bias2_k<<<4, 256, 0, stream>>>(b_ih, b_hh, bsum, 1024);

    // ---- 2. stage inputs (zero-pad + f16 convert) ----
    zero_f16_k<<<cdiv(HB * 3 * 96, 256), 256, 0, stream>>>(SINb, HB * 3 * 96);
    zero_f16_k<<<cdiv(R11 * 64, 256), 256, 0, stream>>>(PIN, R11 * 64);
    zero_f16_k<<<cdiv(R11 * 64, 256), 256, 0, stream>>>(OIN, R11 * 64);
    zero_f16_k<<<cdiv(HB * 352, 256), 256, 0, stream>>>(CATA, HB * 352);
    fill_sin_k<<<cdiv(HB * 48, 256), 256, 0, stream>>>(in_ms, in_ps, in_bs, SINb);
    fill_pair_k<<<cdiv(R11 * 29, 256), 256, 0, stream>>>(in_pst, in_lst, PIN);
    fill_pair_k<<<cdiv(R11 * 29, 256), 256, 0, stream>>>(in_ost, in_rst, OIN);

    // ---- 3. situation embeddings -> catA slots 0..2 ----
    gemmF4(SINb, 96, pk_sit, nullptr, 48, F32T, 64, HB * 3, 3, 4);
    ln(F32T, 64, 48, 48, g_sit, bt_sit, b_match, b_psit, b_ball, 1,
       CATA, 3, 48, 352, HB * 3);

    // ---- 4. state embeddings P = LN([pe;le]), O = LN([oe;re]) ----
    gemmF4(PIN, 64, pk_stP, nullptr, 48, F32T, 64, R11, 2, 4);
    ln(F32T, 64, 48, 64, g_st, bt_st, b_pst, b_lst, nullptr, 2, PIN, 1, 0, 64, R11);
    _Float16* P = PIN;
    gemmF4(OIN, 64, pk_stO, nullptr, 48, F32T, 64, R11, 2, 4);
    ln(F32T, 64, 48, 64, g_st, bt_st, b_ost, b_rst, nullptr, 2, OIN, 1, 0, 64, R11);
    _Float16* O = OIN;

    // ---- 5. two attention stages (batched GEMMs + wave-per-sample softmax) --
    auto stage = [&](const _Float16* Rside, const _Float16* Lside,
                     _Float16* pq1, _Float16* pk1, _Float16* pv1,
                     _Float16* pq2, _Float16* pk2, _Float16* pv2,
                     int prOff, int agOff, float* a1out, float* a2out) {
        gemmH4(Rside, 64, pq1, nullptr, 48, PR0, 64, R11, 2, 4);   // rq1
        gemmH4(Lside, 64, pk1, nullptr, 48, PR1, 64, R11, 2, 4);   // lk1
        gemmH4(Rside, 64, pv1, nullptr, 48, PR2, 64, R11, 2, 4);   // rv1
        gemmH4(Lside, 64, pv1, nullptr, 48, PR3, 64, R11, 2, 4);   // lv1
        att_stage1_k<<<(unsigned)HB, 32, 0, stream>>>(PR0, PR1, PR2, PR3, SUBB, a1out);
        gemmH4(SUBB, 64, pq2, nullptr, 48, PR0, 64, R11, 2, 4);    // sub@q2
        gemmH4(SUBB, 64, pk2, nullptr, 48, PR1, 64, R11, 2, 4);    // sub@k2
        gemmH4(SUBB, 64, pv2, nullptr, 48, PR2, 64, R11, 2, 4);    // sub@v2
        att_stage2_k<<<(unsigned)HB, 32, 0, stream>>>(PR0, PR1, PR2, CATA, prOff, agOff, a2out);
    };
    stage(O, P, pk_q[0], pk_q[1], pk_q[2], pk_q[3], pk_q[4], pk_q[5],
          3 * 48, 4 * 48, dout + O_A1, dout + O_A2);
    stage(P, O, pk_q[6], pk_q[7], pk_q[8], pk_q[9], pk_q[10], pk_q[11],
          5 * 48, 6 * 48, dout + O_D1, dout + O_D2);

    // ---- 6. concat projection + LN -> x (f16) ----
    gemmF4(CATA, 352, pk_cat, nullptr, 256, F32T, 256, HB, 11, 16);
    ln(F32T, 256, 256, 256, g_cat, bt_cat, b_cat, nullptr, nullptr, 0,
       XB, 1, 0, 256, HB);

    // ---- 7. LSTM: 32 steps of [x_t,h] @ [w_ih;w_hh] + bias ----
    lstm_init_k<<<cdiv(BB * 256, 256), 256, 0, stream>>>(in_h0, in_c0, XH, CBUF);
    for (int t = 0; t < HH; ++t) {
        lstm_prep_k<<<cdiv(BB * 256, 256), 256, 0, stream>>>(XB, XH, t);
        gemmF4(XH, 512, pk_lstm, bsum, 1024, F32T, 1024, BB, 16, 64);
        lstm_cell_k<<<cdiv(BB * 256, 256), 256, 0, stream>>>(F32T, CBUF, OUTB, XH, t);
    }
    copy_f32_k<<<cdiv(BB * 256, 256), 256, 0, stream>>>(dout + O_HT,
                                                        OUTB + (size_t)(HH - 1) * BB * 256,
                                                        (long)BB * 256);
    copy_f32_k<<<cdiv(BB * 256, 256), 256, 0, stream>>>(dout + O_CT, CBUF, (long)BB * 256);

    // ---- 8. heads ----
    cvt_f32_f16_k<<<cdiv(HB * 256, 256), 256, 0, stream>>>(OUTB, OUTH, HB * 256);

    // action head
    gemmF4(OUTH, 256, pk_a1, nullptr, 164, F32T, 192, HB, 8, 12);
    ln(F32T, 192, 164, 192, g_a1, bt_a1, b_a1, nullptr, nullptr, 0, AF, 1, 0, 192, HB);
    gemmF1(AF, 192, pk_a2, b_a2, 12, F32T, 16, HB, 6, 1);
    softmax_k<<<cdiv(HB, 256), 256, 0, stream>>>(F32T, 16, 12, in_av, 12,
                                                 dout + O_PROB, 12, HB);
    // move head
    gemmF4(OUTH, 256, pk_m1, nullptr, 164, F32T, 192, HB, 8, 12);
    ln(F32T, 192, 164, 192, g_m1, bt_m1, b_m1, nullptr, nullptr, 0, AF, 1, 0, 192, HB);
    gemmF1(AF, 192, pk_m2, b_m2, 8, F32T, 16, HB, 6, 1);
    softmax_k<<<cdiv(HB, 256), 256, 0, stream>>>(F32T, 16, 8, nullptr, 0,
                                                 dout + O_PROBM, 8, HB);
    // value head
    gemmF4(OUTH, 256, pk_v1, nullptr, 164, F32T, 192, HB, 8, 12);
    ln(F32T, 192, 164, 192, g_v1, bt_v1, b_v1, nullptr, nullptr, 0, AF, 1, 0, 192, HB);
    dotv_k<<<cdiv(HB, 256), 256, 0, stream>>>(AF, w_v2, dout + O_V, HB);
}
